// MultiheadAttention_2697239462071
// MI455X (gfx1250) — compile-verified, matrix operand order checked by reading, not run on hardware
//
#include <hip/hip_runtime.h>

#define S_LEN 2048
#define D_DIM 1024
#define H_NUM 16
#define HDIM  64
#define BATCH 2
#define MROWS (BATCH*S_LEN)   // 4096
#define LDA   72              // padded LDS row stride (elems) for 64-wide rows: 144B, conflict-free

typedef __attribute__((ext_vector_type(16))) __bf16 v16bf;
typedef __attribute__((ext_vector_type(8)))  float  f32x8;
typedef __attribute__((ext_vector_type(4)))  int    v4i;
typedef unsigned short u16;

// ---------------------------------------------------------- async copy to LDS
#if defined(__has_builtin)
# if __has_builtin(__builtin_amdgcn_global_load_async_to_lds_b128)
#  define HAVE_ASYNC_LDS 1
# endif
#endif
#ifndef HAVE_ASYNC_LDS
# define HAVE_ASYNC_LDS 0
#endif

#if HAVE_ASYNC_LDS
typedef __attribute__((address_space(1))) v4i gv4i;   // global
typedef __attribute__((address_space(3))) v4i lv4i;   // LDS
__device__ __forceinline__ void cp_async_b128(u16* lds_dst, const u16* gsrc){
  __builtin_amdgcn_global_load_async_to_lds_b128((gv4i*)gsrc, (lv4i*)lds_dst, 0, 0);
}
__device__ __forceinline__ void cp_async_wait(){
# if __has_builtin(__builtin_amdgcn_s_wait_asynccnt)
  __builtin_amdgcn_s_wait_asynccnt(0);
# else
  asm volatile("s_wait_asynccnt 0x0" ::: "memory");
# endif
}
#else
__device__ __forceinline__ void cp_async_b128(u16* lds_dst, const u16* gsrc){
  *(uint4*)lds_dst = *(const uint4*)gsrc;   // sync fallback; s_wait_dscnt before barrier
}
__device__ __forceinline__ void cp_async_wait(){}
#endif

__device__ __forceinline__ u16 f2bf(float f){
  unsigned int u = __float_as_uint(f);
  u += 0x7FFFu + ((u >> 16) & 1u);       // round-to-nearest-even
  return (u16)(u >> 16);
}

// Load one 16x32 bf16 A-fragment (or mirrored B-fragment).
// ISA layout: lane<16 holds K = kbase+{0..7, 16..23}; lane>=16 holds K = kbase+{8..15, 24..31}.
__device__ __forceinline__ v16bf load_frag(const u16* rowptr, int kbase, int lh){
  union { uint4 q[2]; v16bf v; } u;
  const uint4* p = (const uint4*)(rowptr + kbase + lh*8);
  u.q[0] = p[0];     // 8 bf16 at kbase + lh*8
  u.q[1] = p[2];     // 8 bf16 at kbase + lh*8 + 16
  return u.v;
}

__device__ __forceinline__ f32x8 wmma_bf16(v16bf a, v16bf b, f32x8 c){
  return __builtin_amdgcn_wmma_f32_16x16x32_bf16(false, a, false, b, (short)0, c, false, false);
}

// ---------------------------------------------------------------- fp32 -> bf16
__global__ __launch_bounds__(256) void cvt_bf16_kernel(const float* __restrict__ src,
                                                       u16* __restrict__ dst, int n4){
  int i = blockIdx.x*256 + threadIdx.x;
  if (i < n4){
    float4 f = ((const float4*)src)[i];
    ushort4 o;
    o.x = f2bf(f.x); o.y = f2bf(f.y); o.z = f2bf(f.z); o.w = f2bf(f.w);
    ((ushort4*)dst)[i] = o;
  }
}

// ------------------------------------------------- shared GEMM mainloop (bf16)
// C[128x128 tile] = A[128 x 1024] * B^T  (B row-major [n][k], i.e. torch W)
// 256 threads = 8 waves in 2(M) x 4(N); each wave: 64x32 via 16x16 WMMA tiles.
// K-step 64, double-buffered LDS, async global->LDS staging.
__device__ __forceinline__ void stage_gemm(const u16* __restrict__ Am,
                                           const u16* __restrict__ Bm,
                                           u16* As, u16* Bs,
                                           int by, int bx, int k0, int t){
#pragma unroll
  for (int it = 0; it < 4; ++it){
    int cidx = t + 256*it;           // 1024 chunks of 16B per matrix
    int r  = cidx >> 3;
    int cc = (cidx & 7) << 3;
    cp_async_b128(As + r*LDA + cc, Am + (size_t)(by*128 + r)*D_DIM + k0 + cc);
    cp_async_b128(Bs + r*LDA + cc, Bm + (size_t)(bx*128 + r)*D_DIM + k0 + cc);
  }
}

__device__ __forceinline__ void gemm_mainloop(const u16* __restrict__ Am,
                                              const u16* __restrict__ Bm,
                                              u16* As, u16* Bs,   // each [2][128*LDA]
                                              int by, int bx, f32x8 c[4][2]){
  const int t    = threadIdx.x;
  const int lane = t & 31;
  const int lh   = lane >> 4, lr = lane & 15;
  const int wave = t >> 5;
  const int wm   = wave & 1, wN = wave >> 1;
  const int BUF  = 128*LDA;

  stage_gemm(Am, Bm, As, Bs, by, bx, 0, t);
  for (int k0 = 0; k0 < D_DIM; k0 += 64){
    const int cur = (k0 >> 6) & 1;
    u16* Ac = As + cur*BUF;
    u16* Bc = Bs + cur*BUF;
    cp_async_wait();
    __syncthreads();
    if (k0 + 64 < D_DIM)
      stage_gemm(Am, Bm, As + (cur^1)*BUF, Bs + (cur^1)*BUF, by, bx, k0 + 64, t);
#pragma unroll
    for (int kc = 0; kc < 2; ++kc){
      v16bf a[4], b[2];
#pragma unroll
      for (int i = 0; i < 4; ++i) a[i] = load_frag(Ac + (wm*64 + i*16 + lr)*LDA, kc*32, lh);
#pragma unroll
      for (int j = 0; j < 2; ++j) b[j] = load_frag(Bc + (wN*32 + j*16 + lr)*LDA, kc*32, lh);
#pragma unroll
      for (int i = 0; i < 4; ++i)
#pragma unroll
        for (int j = 0; j < 2; ++j)
          c[i][j] = wmma_bf16(a[i], b[j], c[i][j]);
    }
  }
}

// ------------------------------------------------------------- QKV projection
__global__ __launch_bounds__(256) void gemm_qkv_kernel(
    const u16* __restrict__ xb,
    const u16* __restrict__ wq, const u16* __restrict__ wk, const u16* __restrict__ wv,
    const float* __restrict__ bq, const float* __restrict__ bk, const float* __restrict__ bv,
    u16* __restrict__ qb, u16* __restrict__ kb, u16* __restrict__ vtb){
  __shared__ __align__(16) u16 As[2*128*LDA];
  __shared__ __align__(16) u16 Bs[2*128*LDA];
  const int t    = threadIdx.x;
  const int wave = t >> 5, lane = t & 31;
  const int lh   = lane >> 4, lr = lane & 15;
  const int wm   = wave & 1,  wN = wave >> 1;
  const int bx = blockIdx.x, by = blockIdx.y, w = blockIdx.z;

  const u16*   W    = (w == 0) ? wq : (w == 1) ? wk : wv;
  const float* bias = (w == 0) ? bq : (w == 1) ? bk : bv;

  f32x8 c[4][2];
#pragma unroll
  for (int i = 0; i < 4; ++i)
#pragma unroll
    for (int j = 0; j < 2; ++j)
#pragma unroll
      for (int r = 0; r < 8; ++r) c[i][j][r] = 0.f;

  gemm_mainloop(xb, W, As, Bs, by, bx, c);

#pragma unroll
  for (int i = 0; i < 4; ++i)
#pragma unroll
    for (int j = 0; j < 2; ++j)
#pragma unroll
      for (int r = 0; r < 8; ++r){
        int row = by*128 + wm*64 + i*16 + r + 8*lh;
        int col = bx*128 + wN*32 + j*16 + lr;
        float v = c[i][j][r] + bias[col];
        int bb = row >> 11, ss = row & 2047;
        int hh = col >> 6,  hd = col & 63;
        size_t bh = (size_t)(bb*H_NUM + hh);
        if (w == 0)      qb [(bh*S_LEN + ss)*HDIM + hd] = f2bf(v * 0.125f); // fold 1/sqrt(64)
        else if (w == 1) kb [(bh*S_LEN + ss)*HDIM + hd] = f2bf(v);
        else             vtb[(bh*HDIM + hd)*S_LEN + ss] = f2bf(v);          // V stored transposed
      }
}

// ---------------------------------------------------------- output projection
__global__ __launch_bounds__(256) void gemm_out_kernel(
    const u16* __restrict__ zb, const u16* __restrict__ wo,
    const float* __restrict__ bo, float* __restrict__ out){
  __shared__ __align__(16) u16 As[2*128*LDA];
  __shared__ __align__(16) u16 Bs[2*128*LDA];
  const int t    = threadIdx.x;
  const int wave = t >> 5, lane = t & 31;
  const int lh   = lane >> 4, lr = lane & 15;
  const int wm   = wave & 1,  wN = wave >> 1;
  const int bx = blockIdx.x, by = blockIdx.y;

  f32x8 c[4][2];
#pragma unroll
  for (int i = 0; i < 4; ++i)
#pragma unroll
    for (int j = 0; j < 2; ++j)
#pragma unroll
      for (int r = 0; r < 8; ++r) c[i][j][r] = 0.f;

  gemm_mainloop(zb, wo, As, Bs, by, bx, c);

#pragma unroll
  for (int i = 0; i < 4; ++i)
#pragma unroll
    for (int j = 0; j < 2; ++j)
#pragma unroll
      for (int r = 0; r < 8; ++r){
        int row = by*128 + wm*64 + i*16 + r + 8*lh;
        int col = bx*128 + wN*32 + j*16 + lr;
        out[(size_t)row*D_DIM + col] = c[i][j][r] + bo[col];
      }
}

// --------------------------------------------------- fused causal attention
// Grid: (S/64 q-tiles, B*H). 128 threads = 4 waves; wave owns 16 q rows.
// K/V^T tiles double-buffered in LDS with async prefetch of tile kt+1.
__device__ __forceinline__ void stage_kv(const u16* __restrict__ kb,
                                         const u16* __restrict__ vtb,
                                         u16* Kd, u16* Vd, int bh, int k0, int t){
#pragma unroll
  for (int it = 0; it < 4; ++it){
    int cidx = t + 128*it;           // 512 chunks of 16B per matrix
    int r  = cidx >> 3;
    int cc = (cidx & 7) << 3;
    cp_async_b128(Kd + r*LDA + cc, kb  + ((size_t)bh*S_LEN + k0 + r)*HDIM + cc);
    cp_async_b128(Vd + r*LDA + cc, vtb + ((size_t)bh*HDIM + r)*S_LEN + k0 + cc);
  }
}

__global__ __launch_bounds__(128) void attn_kernel(
    const u16* __restrict__ qb, const u16* __restrict__ kb,
    const u16* __restrict__ vtb, u16* __restrict__ zb){
  __shared__ __align__(16) u16 Ks[2*64*LDA];
  __shared__ __align__(16) u16 Vs[2*64*LDA];
  __shared__ __align__(16) u16 Ps[4*16*LDA];
  const int qt = blockIdx.x, bh = blockIdx.y;
  const int t = threadIdx.x, wave = t >> 5, lane = t & 31;
  const int lh = lane >> 4, lr = lane & 15;
  const int q0w = qt*64 + wave*16;
  const int BUF = 64*LDA;
  const float LOG2E = 1.4426950408889634f;

  // Q fragments held in registers for the whole pass (Q pre-scaled by 1/8)
  const u16* Qrow = qb + ((size_t)bh*S_LEN + q0w + lr)*HDIM;
  v16bf qf0 = load_frag(Qrow, 0, lh);
  v16bf qf1 = load_frag(Qrow, 32, lh);

  f32x8 o[4];
#pragma unroll
  for (int g = 0; g < 4; ++g)
#pragma unroll
    for (int r = 0; r < 8; ++r) o[g][r] = 0.f;
  float m[8], l[8];
#pragma unroll
  for (int r = 0; r < 8; ++r){ m[r] = -1e30f; l[r] = 0.f; }

  stage_kv(kb, vtb, Ks, Vs, bh, 0, t);
  for (int kt = 0; kt <= qt; ++kt){
    const int k0  = kt*64;
    const int cur = kt & 1;
    u16* Kc = Ks + cur*BUF;
    u16* Vc = Vs + cur*BUF;
    cp_async_wait();
    __syncthreads();
    if (kt < qt)
      stage_kv(kb, vtb, Ks + (cur^1)*BUF, Vs + (cur^1)*BUF, bh, k0 + 64, t);

    // S = Q K^T   (16 x 64 per wave)
    f32x8 s[4];
#pragma unroll
    for (int g = 0; g < 4; ++g){
#pragma unroll
      for (int r = 0; r < 8; ++r) s[g][r] = 0.f;
      v16bf b0 = load_frag(Kc + (g*16 + lr)*LDA, 0,  lh);
      v16bf b1 = load_frag(Kc + (g*16 + lr)*LDA, 32, lh);
      s[g] = wmma_bf16(qf0, b0, s[g]);
      s[g] = wmma_bf16(qf1, b1, s[g]);
    }
    // causal mask
#pragma unroll
    for (int g = 0; g < 4; ++g){
      int col = k0 + g*16 + lr;
#pragma unroll
      for (int r = 0; r < 8; ++r){
        int row = q0w + r + 8*lh;
        if (col > row) s[g][r] = -1e30f;
      }
    }
    // row max + online rescale (16-lane half-wave reductions)
#pragma unroll
    for (int r = 0; r < 8; ++r){
      float v = fmaxf(fmaxf(s[0][r], s[1][r]), fmaxf(s[2][r], s[3][r]));
      v = fmaxf(v, __shfl_xor(v, 1, 32));
      v = fmaxf(v, __shfl_xor(v, 2, 32));
      v = fmaxf(v, __shfl_xor(v, 4, 32));
      v = fmaxf(v, __shfl_xor(v, 8, 32));
      float mn = fmaxf(m[r], v);
      float sc = exp2f((m[r] - mn) * LOG2E);
      m[r] = mn;
      l[r] *= sc;
#pragma unroll
      for (int g = 0; g < 4; ++g) o[g][r] *= sc;
    }
    // p = exp(s - m), row sums
    float rs[8];
#pragma unroll
    for (int r = 0; r < 8; ++r) rs[r] = 0.f;
#pragma unroll
    for (int g = 0; g < 4; ++g)
#pragma unroll
      for (int r = 0; r < 8; ++r){
        float p = exp2f((s[g][r] - m[r]) * LOG2E);
        s[g][r] = p;
        rs[r] += p;
      }
#pragma unroll
    for (int r = 0; r < 8; ++r){
      float v = rs[r];
      v += __shfl_xor(v, 1, 32);
      v += __shfl_xor(v, 2, 32);
      v += __shfl_xor(v, 4, 32);
      v += __shfl_xor(v, 8, 32);
      l[r] += v;
    }
    // stage P (bf16) into wave-private LDS to re-lay C-layout -> A-layout
    u16* Pw = Ps + wave*16*LDA;
#pragma unroll
    for (int g = 0; g < 4; ++g)
#pragma unroll
      for (int r = 0; r < 8; ++r)
        Pw[(r + 8*lh)*LDA + g*16 + lr] = f2bf(s[g][r]);
    // O += P @ V   (Vs holds V^T: row d over kk = exactly the B-fragment shape)
#pragma unroll
    for (int kc = 0; kc < 2; ++kc){
      v16bf pf = load_frag(Pw + lr*LDA, kc*32, lh);
#pragma unroll
      for (int g = 0; g < 4; ++g){
        v16bf vf = load_frag(Vc + (g*16 + lr)*LDA, kc*32, lh);
        o[g] = wmma_bf16(pf, vf, o[g]);
      }
    }
  }

  // epilogue: normalize and write Z [B,S,D] bf16
  int b = bh >> 4, h = bh & 15;
#pragma unroll
  for (int g = 0; g < 4; ++g)
#pragma unroll
    for (int r = 0; r < 8; ++r){
      int row = q0w + r + 8*lh;
      int col = h*HDIM + g*16 + lr;
      zb[((size_t)b*S_LEN + row)*D_DIM + col] = f2bf(o[g][r] / l[r]);
    }
}

// ---------------------------------------------------------------------- launch
extern "C" void kernel_launch(void* const* d_in, const int* in_sizes, int n_in,
                              void* d_out, int out_size, void* d_ws, size_t ws_size,
                              hipStream_t stream){
  (void)in_sizes; (void)n_in; (void)out_size; (void)ws_size;
  const float* x  = (const float*)d_in[0];
  const float* Wq = (const float*)d_in[1];
  const float* bq = (const float*)d_in[2];
  const float* Wk = (const float*)d_in[3];
  const float* bk = (const float*)d_in[4];
  const float* Wv = (const float*)d_in[5];
  const float* bv = (const float*)d_in[6];
  const float* Wo = (const float*)d_in[7];
  const float* bo = (const float*)d_in[8];
  float* out = (float*)d_out;

  char* ws = (char*)d_ws;
  const size_t MB = 1024*1024;
  u16* xb  = (u16*)(ws);             // 8 MB  x in bf16
  u16* wqb = (u16*)(ws + 8*MB);      // 2 MB each weight
  u16* wkb = (u16*)(ws + 10*MB);
  u16* wvb = (u16*)(ws + 12*MB);
  u16* wob = (u16*)(ws + 14*MB);
  u16* qb  = (u16*)(ws + 16*MB);     // 8 MB  Q [B,H,S,HD]
  u16* kb  = (u16*)(ws + 24*MB);     // 8 MB  K [B,H,S,HD]
  u16* vtb = (u16*)(ws + 32*MB);     // 8 MB  V^T [B,H,HD,S]
  u16* zb  = (u16*)(ws + 40*MB);     // 8 MB  Z [B,S,D]

  cvt_bf16_kernel<<<4096, 256, 0, stream>>>(x,  xb,  MROWS*D_DIM/4);
  cvt_bf16_kernel<<<1024, 256, 0, stream>>>(Wq, wqb, D_DIM*D_DIM/4);
  cvt_bf16_kernel<<<1024, 256, 0, stream>>>(Wk, wkb, D_DIM*D_DIM/4);
  cvt_bf16_kernel<<<1024, 256, 0, stream>>>(Wv, wvb, D_DIM*D_DIM/4);
  cvt_bf16_kernel<<<1024, 256, 0, stream>>>(Wo, wob, D_DIM*D_DIM/4);

  gemm_qkv_kernel<<<dim3(8, 32, 3), 256, 0, stream>>>(xb, wqb, wkb, wvb,
                                                      bq, bk, bv, qb, kb, vtb);
  attn_kernel<<<dim3(32, 32), 128, 0, stream>>>(qb, kb, vtb, zb);
  gemm_out_kernel<<<dim3(8, 32), 256, 0, stream>>>(zb, wob, bo, out);
}
